// GCAModule_86062554677632
// MI455X (gfx1250) — matrix operand order repaired and checked by reading, not compiled, not run on hardware
//
#include <hip/hip_runtime.h>

typedef __attribute__((ext_vector_type(16))) _Float16 v16h;
typedef __attribute__((ext_vector_type(8)))  _Float16 v8h;
typedef __attribute__((ext_vector_type(8)))  float    v8f;

#define NB   4
#define CIN  256
#define C2   128
#define ACH  128
#define HL   32
#define LL   1024      // 32*32 low-res positions
#define KP   1152      // 128 * 9 patch length
#define M2   2048      // 128 * 16 alpha-patch length
#define SP   4096      // 64*64
#define NSP  16384     // 4 * 4096

#define LROW 72        // LDS panel row stride in halfs (64 data + 8 pad = 144B)

// ---------------- workspace layout (bytes, all 256-aligned) ----------------
// region0 [0,32MB): {X,Wg,glow,P,R} early, then REC (all dead before REC)
#define OFF_X      ((size_t)0)            // f16 [4][1024][256]  = 2 MB
#define OFF_WG     ((size_t)2097152)      // f16 [128][256]      = 64 KB
#define OFF_GLOW   ((size_t)2162688)      // f32 [4][128][1024]  = 2 MB
#define OFF_P      ((size_t)4259840)      // f16 [4][1024][1152] = 9 MB
#define OFF_R      ((size_t)13697024)     // f32 [4][1024][1024] = 16 MB
#define OFF_REC    ((size_t)0)            // f32 [4][1024][2048] = 32 MB (reuse)
#define OFF_SM     ((size_t)33554432)     // f16 [4][1024][1024] = 8 MB
#define OFF_APT    ((size_t)41943040)     // f16 [4][2048][1024] = 16 MB
#define OFF_PROPT  ((size_t)33554432)     // f16 [16384][128]    = 4 MB (reuse SM)
#define OFF_YCONV  ((size_t)37748736)     // f32 [128][16384]    = 8 MB (reuse SM tail + APt head)
#define OFF_WO     ((size_t)58720256)     // f16 [128][128]
#define OFF_ULOW   ((size_t)58753024)     // f32 [4][1024]
#define OFF_UMSC   ((size_t)58769408)     // f32 [4][2]
#define OFF_CSC    ((size_t)58769664)     // f32 [4][1024]
#define OFF_PEN    ((size_t)58786048)     // f32 [4][1024]
#define OFF_INVN   ((size_t)58802432)     // f32 [4][1024]
#define OFF_STATS  ((size_t)58818816)     // f32 [128][2]

__device__ __forceinline__ int reflect_i(int i, int n) {
  if (i < 0) i = -i;
  if (i >= n) i = 2 * n - 2 - i;
  return i;
}

// ---------------- WMMA fragment loads from LDS panels ----------------
// A 16x32 f16 (ISA 7.12.2): lane<16 -> row=lane, K {ks..ks+7} then {ks+16..ks+23}
//                           lane>=16 -> row=lane-16, K {ks+8..}/{ks+24..}
__device__ __forceinline__ v16h lds_fragA(const _Float16* sm, int row, int ks, int lane) {
  const _Float16* p = sm + row * LROW + ks + ((lane & 16) ? 8 : 0);
  v8h lo = *(const v8h*)p;
  v8h hi = *(const v8h*)(p + 16);
  v16h f;
#pragma unroll
  for (int i = 0; i < 8; ++i) { f[i] = lo[i]; f[i + 8] = hi[i]; }
  return f;
}
// B 32x16 f16: lane<16 -> col=lane holds K {ks..ks+7} then {ks+16..ks+23}; lane>=16 -> K {ks+8..}/{ks+24..}
__device__ __forceinline__ v16h lds_fragB(const _Float16* sm, int row, int ks, int lane) {
  const _Float16* p = sm + row * LROW + ks + ((lane & 16) ? 8 : 0);
  v8h lo = *(const v8h*)p;
  v8h hi = *(const v8h*)(p + 16);
  v16h f;
#pragma unroll
  for (int i = 0; i < 8; ++i) { f[i] = lo[i]; f[i + 8] = hi[i]; }
  return f;
}

// ---------------- LDS-tiled GEMM: D[M][N] = A[M][K] * Bt[N][K]^T (+bias[row]) ----
// macro tile 128x128 per 256-thread workgroup; 8 waves each compute 32x64.
// A, Bt row-major over K (f16); D f32.
__global__ __launch_bounds__(256) void k_gemm_lds(
    const _Float16* __restrict__ A, const _Float16* __restrict__ Bt,
    float* __restrict__ D,
    int lda, int ldb, int ldd, int K,
    int tilesM, int tilesN,
    size_t strideA, size_t strideB, size_t strideD,
    const float* __restrict__ bias) {
  __shared__ _Float16 smA[128 * LROW];   // 18 KB
  __shared__ _Float16 smB[128 * LROW];   // 18 KB

  const int perB = tilesM * tilesN;
  const int blk = blockIdx.x;
  const int n = blk / perB;
  const int rem = blk % perB;
  const int tm = rem / tilesN, tn = rem % tilesN;

  const _Float16* Ab = A + (size_t)n * strideA + (size_t)(tm * 128) * lda;
  const _Float16* Bb = Bt + (size_t)n * strideB + (size_t)(tn * 128) * ldb;
  float* Db = D + (size_t)n * strideD;
  const int r0g = tm * 128, c0g = tn * 128;

  const int t = threadIdx.x;
  const int lane = t & 31, w = t >> 5;
  const int wr = w >> 1, wc = w & 1;       // 4x2 wave grid -> 32x64 per wave
  const int lr = lane & 15;

  v8f acc[2][4];
#pragma unroll
  for (int i = 0; i < 2; ++i)
#pragma unroll
    for (int j = 0; j < 4; ++j) acc[i][j] = (v8f){};

  for (int k0 = 0; k0 < K; k0 += 64) {
    // ---- stage 128x64 panels of A and B into LDS (4 16B chunks/thread/panel)
#pragma unroll
    for (int i = 0; i < 4; ++i) {
      int ch = t + (i << 8);               // 0..1023
      int row = ch >> 3, c16 = ch & 7;
      v8h va = *(const v8h*)(Ab + (size_t)row * lda + k0 + c16 * 8);
      v8h vb = *(const v8h*)(Bb + (size_t)row * ldb + k0 + c16 * 8);
      *(v8h*)(smA + row * LROW + c16 * 8) = va;
      *(v8h*)(smB + row * LROW + c16 * 8) = vb;
    }
    __syncthreads();

    // ---- prefetch next k-panels (64B granules; 128 rows x 2 halves = 256 thr)
    if (k0 + 64 < K) {
      int row = t >> 1, half = (t & 1) * 32;
      __builtin_prefetch(Ab + (size_t)row * lda + k0 + 64 + half, 0, 3);
      __builtin_prefetch(Bb + (size_t)row * ldb + k0 + 64 + half, 0, 3);
    }

    // ---- compute: two 16x16x32 k-substeps
#pragma unroll
    for (int ks = 0; ks < 64; ks += 32) {
      v16h a0 = lds_fragA(smA, wr * 32 + lr, ks, lane);
      v16h a1 = lds_fragA(smA, wr * 32 + 16 + lr, ks, lane);
      v16h b0 = lds_fragB(smB, wc * 64 + lr, ks, lane);
      v16h b1 = lds_fragB(smB, wc * 64 + 16 + lr, ks, lane);
      v16h b2 = lds_fragB(smB, wc * 64 + 32 + lr, ks, lane);
      v16h b3 = lds_fragB(smB, wc * 64 + 48 + lr, ks, lane);
      acc[0][0] = __builtin_amdgcn_wmma_f32_16x16x32_f16(false, a0, false, b0, (short)0, acc[0][0], false, false);
      acc[0][1] = __builtin_amdgcn_wmma_f32_16x16x32_f16(false, a0, false, b1, (short)0, acc[0][1], false, false);
      acc[0][2] = __builtin_amdgcn_wmma_f32_16x16x32_f16(false, a0, false, b2, (short)0, acc[0][2], false, false);
      acc[0][3] = __builtin_amdgcn_wmma_f32_16x16x32_f16(false, a0, false, b3, (short)0, acc[0][3], false, false);
      acc[1][0] = __builtin_amdgcn_wmma_f32_16x16x32_f16(false, a1, false, b0, (short)0, acc[1][0], false, false);
      acc[1][1] = __builtin_amdgcn_wmma_f32_16x16x32_f16(false, a1, false, b1, (short)0, acc[1][1], false, false);
      acc[1][2] = __builtin_amdgcn_wmma_f32_16x16x32_f16(false, a1, false, b2, (short)0, acc[1][2], false, false);
      acc[1][3] = __builtin_amdgcn_wmma_f32_16x16x32_f16(false, a1, false, b3, (short)0, acc[1][3], false, false);
    }
    __syncthreads();
  }

  // ---- store 32x64 per wave: D layout: lane&15=col, (lane>>4)*8+v = row
  const int col = lane & 15;
  const int ro = (lane >> 4) * 8;
#pragma unroll
  for (int i = 0; i < 2; ++i) {
#pragma unroll
    for (int v = 0; v < 8; ++v) {
      int row = r0g + wr * 32 + i * 16 + ro + v;
      float bv = bias ? bias[row] : 0.0f;
      float* dr = Db + (size_t)row * ldd + c0g + wc * 64 + col;
#pragma unroll
      for (int j = 0; j < 4; ++j) dr[j * 16] = acc[i][j][v] + bv;
    }
  }
}

// ---------------- other kernels ----------------
__global__ __launch_bounds__(256) void k_convert_w(const float* gw, const float* ow,
                                                   _Float16* Wg, _Float16* Wo) {
  int idx = blockIdx.x * 256 + threadIdx.x;  // 49152 total
  if (idx < 32768) Wg[idx] = (_Float16)gw[idx];
  else             Wo[idx - 32768] = (_Float16)ow[idx - 32768];
}

__global__ __launch_bounds__(256) void k_build_X(const float* img, _Float16* X) {
  int idx = blockIdx.x * 256 + threadIdx.x;  // ((n*1024+s)*256+c), 1,048,576 total
  int c = idx & 255, s = (idx >> 8) & 1023, n = idx >> 18;
  int sy = s >> 5, sx = s & 31;
  X[idx] = (_Float16)img[(((size_t)n * CIN + c) << 12) + sy * 128 + sx * 2];
}

__global__ __launch_bounds__(256) void k_scales(const float* unknown, float* u_low, float* umsc) {
  __shared__ float red[256];
  int n = blockIdx.x, t = threadIdx.x;
  float s = 0.0f;
  for (int i = t; i < LL; i += 256) {
    int sy = i >> 5, sx = i & 31;
    float v = unknown[(size_t)n * SP + sy * 128 + sx * 2];
    u_low[n * LL + i] = v;
    s += v;
  }
  red[t] = s; __syncthreads();
  for (int st = 128; st > 0; st >>= 1) { if (t < st) red[t] += red[t + st]; __syncthreads(); }
  if (t == 0) {
    float um = red[0] * (1.0f / 1024.0f);
    float km = 1.0f - um;
    float us = fminf(fmaxf(sqrtf(um / km), 0.1f), 10.0f);
    float ks = fminf(fmaxf(sqrtf(km / um), 0.1f), 10.0f);
    umsc[n * 2] = us; umsc[n * 2 + 1] = ks;
  }
}

__global__ __launch_bounds__(256) void k_colscale(const float* u_low, const float* umsc,
                                                  float* csc, float* pen) {
  int idx = blockIdx.x * 256 + threadIdx.x;  // 4096
  int n = idx >> 10, p = idx & 1023;
  int py = p >> 5, px = p & 31;
  const float* u = u_low + (n << 10);
  float s = 0.0f;
  for (int dy = -1; dy <= 1; ++dy) {
    int ry = reflect_i(py + dy, 32) * 32;
    for (int dx = -1; dx <= 1; ++dx) s += u[ry + reflect_i(px + dx, 32)];
  }
  float ups = s * (1.0f / 9.0f);
  csc[idx] = (ups > 0.0f) ? umsc[n * 2] : umsc[n * 2 + 1];
  pen[idx] = -10000.0f * ups;
}

__global__ __launch_bounds__(256) void k_build_P(const float* glow, _Float16* P) {
  int idx = blockIdx.x * 256 + threadIdx.x;  // (n*1024+p)*128+c, 524288 total
  int c = idx & 127, p = (idx >> 7) & 1023, n = idx >> 17;
  int py = p >> 5, px = p & 31;
  const float* g = glow + ((size_t)(n * C2 + c) << 10);
  _Float16* dst = P + (size_t)((n << 10) + p) * KP + c * 9;
  int t = 0;
  for (int dy = -1; dy <= 1; ++dy) {
    int ry = reflect_i(py + dy, 32) * 32;
    for (int dx = -1; dx <= 1; ++dx) dst[t++] = (_Float16)g[ry + reflect_i(px + dx, 32)];
  }
}

__global__ __launch_bounds__(256) void k_invnorm(const float* glow, float* invn) {
  int idx = blockIdx.x * 256 + threadIdx.x;  // 4096
  int n = idx >> 10, q = idx & 1023;
  int qy = q >> 5, qx = q & 31;
  int ry[3], rx[3];
  for (int d = 0; d < 3; ++d) { ry[d] = reflect_i(qy + d - 1, 32) * 32; rx[d] = reflect_i(qx + d - 1, 32); }
  const float* g = glow + ((size_t)(n * C2) << 10);
  float s = 0.0f;
  for (int c = 0; c < C2; ++c) {
    const float* gc = g + (c << 10);
    for (int a = 0; a < 3; ++a)
      for (int b = 0; b < 3; ++b) { float v = gc[ry[a] + rx[b]]; s += v * v; }
  }
  invn[idx] = 1.0f / fmaxf(sqrtf(s), 1e-4f);
}

__global__ __launch_bounds__(256) void k_softmax(const float* R, const float* invn,
                                                 const float* csc, const float* pen,
                                                 _Float16* SM) {
  __shared__ float red[256];
  int n = blockIdx.x >> 10, p = blockIdx.x & 1023;
  const float* row = R + ((size_t)(n << 10) + p) * LL;
  const float* iq = invn + (n << 10);
  const float* sq = csc + (n << 10);
  float pdiag = pen[(n << 10) + p];
  int t = threadIdx.x;
  float vals[4];
  float m = -3.4e38f;
#pragma unroll
  for (int i = 0; i < 4; ++i) {
    int q = t + (i << 8);
    float v = row[q] * iq[q] * sq[q];
    if (q == p) v += pdiag;
    vals[i] = v;
    m = fmaxf(m, v);
  }
  red[t] = m; __syncthreads();
  for (int s = 128; s > 0; s >>= 1) { if (t < s) red[t] = fmaxf(red[t], red[t + s]); __syncthreads(); }
  m = red[0]; __syncthreads();
  float sum = 0.0f;
#pragma unroll
  for (int i = 0; i < 4; ++i) { vals[i] = __expf(vals[i] - m); sum += vals[i]; }
  red[t] = sum; __syncthreads();
  for (int s = 128; s > 0; s >>= 1) { if (t < s) red[t] += red[t + s]; __syncthreads(); }
  float inv = 1.0f / red[0];
  _Float16* out = SM + ((size_t)(n << 10) + p) * LL;
#pragma unroll
  for (int i = 0; i < 4; ++i) out[t + (i << 8)] = (_Float16)(vals[i] * inv);
}

__global__ __launch_bounds__(256) void k_build_APt(const float* alpha, _Float16* APt) {
  int idx = blockIdx.x * 256 + threadIdx.x;  // (n*2048+m)*1024+q, 8,388,608 total
  int q = idx & 1023, m = (idx >> 10) & 2047, n = idx >> 21;
  int c = m >> 4, ky = (m >> 2) & 3, kx = m & 3;
  int qy = q >> 5, qx = q & 31;
  int ry = reflect_i(2 * qy + ky - 1, 64);
  int rx = reflect_i(2 * qx + kx - 1, 64);
  APt[idx] = (_Float16)alpha[(((size_t)n * ACH + c) << 12) + ry * 64 + rx];
}

__global__ __launch_bounds__(256) void k_overlap(const float* REC, _Float16* propT) {
  int idx = blockIdx.x * 256 + threadIdx.x;  // n*128*4096 + c*4096 + yx, 2,097,152 total
  int yx = idx & 4095, c = (idx >> 12) & 127, n = idx >> 19;
  int Y = yx >> 6, X = yx & 63;
  const float* r = REC + ((size_t)n << 10) * M2 + c * 16;
  float acc = 0.0f;
  int ky0 = (Y & 1) ? 0 : 1;
  int kx0 = (X & 1) ? 0 : 1;
  for (int ky = ky0; ky < 4; ky += 2) {
    int y = (Y + 1 - ky) >> 1;
    if ((unsigned)y >= 32u) continue;
    for (int kx = kx0; kx < 4; kx += 2) {
      int x = (X + 1 - kx) >> 1;
      if ((unsigned)x >= 32u) continue;
      acc += r[(size_t)(y * 32 + x) * M2 + ky * 4 + kx];
    }
  }
  propT[((size_t)(n << 12) + yx) * 128 + c] = (_Float16)(acc * 0.25f);
}

__global__ __launch_bounds__(256) void k_stats(const float* yconv, float* stats) {
  __shared__ float r1[256], r2[256];
  int o = blockIdx.x, t = threadIdx.x;
  const float* row = yconv + (size_t)o * NSP;
  float s = 0.0f, s2 = 0.0f;
  for (int i = t; i < NSP; i += 256) { float v = row[i]; s += v; s2 += v * v; }
  r1[t] = s; r2[t] = s2; __syncthreads();
  for (int st = 128; st > 0; st >>= 1) {
    if (t < st) { r1[t] += r1[t + st]; r2[t] += r2[t + st]; }
    __syncthreads();
  }
  if (t == 0) {
    float mu = r1[0] * (1.0f / 16384.0f);
    float var = fmaxf(r2[0] * (1.0f / 16384.0f) - mu * mu, 0.0f);
    stats[o * 2] = mu;
    stats[o * 2 + 1] = rsqrtf(var + 1e-5f);
  }
}

__global__ __launch_bounds__(256) void k_final(const float* yconv, const float* stats,
                                               const float* gamma, const float* beta,
                                               const float* alpha, float* out) {
  int idx = blockIdx.x * 256 + threadIdx.x;  // (n,o,yx) flat, 2,097,152 total
  int yx = idx & 4095, o = (idx >> 12) & 127, n = idx >> 19;
  float v = yconv[(size_t)o * NSP + (n << 12) + yx];
  v = (v - stats[o * 2]) * stats[o * 2 + 1] * gamma[o] + beta[o];
  out[idx] = v + alpha[idx];
}

// ---------------- launch ----------------
extern "C" void kernel_launch(void* const* d_in, const int* in_sizes, int n_in,
                              void* d_out, int out_size, void* d_ws, size_t ws_size,
                              hipStream_t stream) {
  const float* img_feat   = (const float*)d_in[0];
  const float* alpha_feat = (const float*)d_in[1];
  const float* unknown    = (const float*)d_in[2];
  const float* gconv_w    = (const float*)d_in[3];
  const float* gconv_b    = (const float*)d_in[4];
  const float* oconv_w    = (const float*)d_in[5];
  const float* bn_gamma   = (const float*)d_in[6];
  const float* bn_beta    = (const float*)d_in[7];
  float* out = (float*)d_out;

  char* w = (char*)d_ws;
  _Float16* X     = (_Float16*)(w + OFF_X);
  _Float16* Wg    = (_Float16*)(w + OFF_WG);
  float*    glow  = (float*)   (w + OFF_GLOW);
  _Float16* P     = (_Float16*)(w + OFF_P);
  float*    R     = (float*)   (w + OFF_R);
  float*    REC   = (float*)   (w + OFF_REC);
  _Float16* SM    = (_Float16*)(w + OFF_SM);
  _Float16* APt   = (_Float16*)(w + OFF_APT);
  _Float16* propT = (_Float16*)(w + OFF_PROPT);
  float*    yconv = (float*)   (w + OFF_YCONV);
  _Float16* Wo    = (_Float16*)(w + OFF_WO);
  float*    u_low = (float*)   (w + OFF_ULOW);
  float*    umsc  = (float*)   (w + OFF_UMSC);
  float*    csc   = (float*)   (w + OFF_CSC);
  float*    pen   = (float*)   (w + OFF_PEN);
  float*    invn  = (float*)   (w + OFF_INVN);
  float*    stats = (float*)   (w + OFF_STATS);

  k_convert_w<<<192, 256, 0, stream>>>(gconv_w, oconv_w, Wg, Wo);
  k_build_X  <<<4096, 256, 0, stream>>>(img_feat, X);
  // glow = Wg(128x256) * X^T : M=128, N=1024, K=256, 4 batches (A shared)
  k_gemm_lds <<<32, 256, 0, stream>>>(Wg, X, glow, CIN, CIN, LL, CIN, 1, 8,
                                      (size_t)0, (size_t)LL * CIN, (size_t)C2 * LL, gconv_b);
  k_scales   <<<NB, 256, 0, stream>>>(unknown, u_low, umsc);
  k_colscale <<<16, 256, 0, stream>>>(u_low, umsc, csc, pen);
  k_build_P  <<<2048, 256, 0, stream>>>(glow, P);
  k_invnorm  <<<16, 256, 0, stream>>>(glow, invn);
  // R = P * P^T : M=N=1024, K=1152, 4 batches
  k_gemm_lds <<<256, 256, 0, stream>>>(P, P, R, KP, KP, LL, KP, 8, 8,
                                       (size_t)LL * KP, (size_t)LL * KP, (size_t)LL * LL, nullptr);
  k_softmax  <<<4096, 256, 0, stream>>>(R, invn, csc, pen, SM);
  k_build_APt<<<32768, 256, 0, stream>>>(alpha_feat, APt);
  // REC = SM * APt^T : M=1024, N=2048, K=1024, 4 batches
  k_gemm_lds <<<512, 256, 0, stream>>>(SM, APt, REC, LL, LL, M2, LL, 8, 16,
                                       (size_t)LL * LL, (size_t)M2 * LL, (size_t)LL * M2, nullptr);
  k_overlap  <<<8192, 256, 0, stream>>>(REC, propT);
  // yconv = Wo(128x128) * propT^T : M=128, N=16384, K=128, 1 batch
  k_gemm_lds <<<128, 256, 0, stream>>>(Wo, propT, yconv, ACH, ACH, NSP, ACH, 1, 128,
                                       (size_t)0, (size_t)0, (size_t)0, nullptr);
  k_stats    <<<128, 256, 0, stream>>>(yconv, stats);
  k_final    <<<8192, 256, 0, stream>>>(yconv, stats, bn_gamma, bn_beta, alpha_feat, out);
}
